// fwPKM_17978733101788
// MI455X (gfx1250) — matrix operand order repaired and checked by reading, not compiled, not run on hardware
//
#include <hip/hip_runtime.h>
#include <math.h>

#define DIM      1024
#define HEADS    4
#define NUM_KEYS 512
#define DQK      128
#define DV       128
#define TOPK     8
#define NTOK     4096            // b*n = 2*2048
#define IDW_EPS  0.001f
#define RMS_EPS  1.1920929e-07f
#define BIGF     3.0e38f

typedef __attribute__((ext_vector_type(16))) __bf16 v16bf;
typedef __attribute__((ext_vector_type(8)))  __bf16 v8bf;
typedef __attribute__((ext_vector_type(8)))  float  v8f;
typedef __attribute__((ext_vector_type(4)))  float  v4f;

// ---------- helpers ----------
__device__ inline unsigned short f2bf(float f) {
    union { float f; unsigned int u; } v; v.f = f;
    unsigned int u = v.u;
    u += 0x7FFFu + ((u >> 16) & 1u);          // round-to-nearest-even
    return (unsigned short)(u >> 16);
}
__device__ inline float bf2f(unsigned short h) {
    union { unsigned int u; float f; } v; v.u = ((unsigned int)h) << 16;
    return v.f;
}
__device__ inline float wave_sum(float v) {
    #pragma unroll
    for (int off = 16; off; off >>= 1) v += __shfl_xor(v, off, 32);
    return v;
}
__device__ inline float block_sum(float v, float* sred) {
    int lane = threadIdx.x & 31, wid = threadIdx.x >> 5, nw = (int)(blockDim.x >> 5);
    v = wave_sum(v);
    if (lane == 0) sred[wid] = v;
    __syncthreads();
    if (threadIdx.x == 0) {
        float t = 0.f;
        for (int i = 0; i < nw; ++i) t += sred[i];
        sred[0] = t;
    }
    __syncthreads();
    float r = sred[0];
    __syncthreads();
    return r;
}

// A fragment (16x32, MxK), row-major source, per ISA 7.12.2:
// lanes 0-15: row m0+lane, K = k0+[0..7] and k0+[16..23]
// lanes 16-31: row m0+lane-16, K = k0+[8..15] and k0+[24..31]
__device__ inline v16bf frag_a(const unsigned short* A, int ld, int m0, int k0, int lane) {
    int row = m0 + (lane & 15);
    int kb  = k0 + ((lane >> 4) << 3);
    const __bf16* base = (const __bf16*)(A + (size_t)row * ld + kb);
    v8bf lo = *(const v8bf*)base;
    v8bf hi = *(const v8bf*)(base + 16);
    v16bf f;
    #pragma unroll
    for (int i = 0; i < 8; ++i) { f[i] = lo[i]; f[i + 8] = hi[i]; }
    return f;
}
// B fragment (32x16, KxN) with B[k][n] = W[n][k], W row-major (N x K):
// lanes 0-15: col n0+lane, K = k0+[0..7] and k0+[16..23]; lanes 16-31: col n0+lane-16, K = k0+[8..15] and k0+[24..31]
__device__ inline v16bf frag_b(const unsigned short* W, int ld, int n0, int k0, int lane) {
    int col = n0 + (lane & 15);
    int kb  = k0 + ((lane >> 4) << 3);
    const __bf16* base = (const __bf16*)(W + (size_t)col * ld + kb);
    v8bf lo = *(const v8bf*)base;
    v8bf hi = *(const v8bf*)(base + 16);
    v16bf f;
    #pragma unroll
    for (int i = 0; i < 8; ++i) { f[i] = lo[i]; f[i + 8] = hi[i]; }
    return f;
}

// ---------- kernel 1: fold norm weights into bf16 weight copies ----------
__global__ void k_prep_weights(const float* __restrict__ Wq, const float* __restrict__ qw,
                               const float* __restrict__ Wv, const float* __restrict__ vw,
                               const float* __restrict__ Wo, const float* __restrict__ ow,
                               const float* __restrict__ keys, const float* __restrict__ Wg,
                               const float* __restrict__ gw,
                               unsigned short* __restrict__ Wqb, unsigned short* __restrict__ Wvb,
                               unsigned short* __restrict__ Wob, unsigned short* __restrict__ Kb,
                               float* __restrict__ Wgs) {
    int id = blockIdx.x * blockDim.x + threadIdx.x;
    const int nQ = 1024 * 1024, nV = 512 * 1024, nO = 1024 * 512, nK = 8 * 512 * 128, nG = 1024;
    if (id < nQ) { Wqb[id] = f2bf(Wq[id] * qw[id & 1023]); return; }
    id -= nQ;
    if (id < nV) { Wvb[id] = f2bf(Wv[id] * vw[id & 1023]); return; }
    id -= nV;
    if (id < nO) { Wob[id] = f2bf(Wo[id] * ow[id & 511]); return; }
    id -= nO;
    if (id < nK) { Kb[id] = f2bf(keys[id]); return; }
    id -= nK;
    if (id < nG) { Wgs[id] = Wg[id] * gw[id]; }
}

// ---------- kernel 2: key squared norms ----------
__global__ void k_kss(const float* __restrict__ keys, float* __restrict__ kss) {
    int id = blockIdx.x * blockDim.x + threadIdx.x;   // hh*512+key, 4096 total
    if (id >= 8 * NUM_KEYS) return;
    const float* p = keys + (size_t)id * DQK;
    float s = 0.f;
    #pragma unroll 4
    for (int t = 0; t < DQK; ++t) s += p[t] * p[t];
    kss[id] = s;
}

// ---------- kernel 3: token rmsnorm (bf16) + gate ----------
__global__ void k_norm_tokens(const float* __restrict__ tokens, const float* __restrict__ Wgs,
                              unsigned short* __restrict__ Xn, float* __restrict__ gate) {
    __shared__ float sred[8];
    __shared__ float sInv;
    int tok = blockIdx.x, tid = threadIdx.x;          // 256 threads, 4 elems each
    const float* x = tokens + (size_t)tok * DIM;
    v4f a = *(const v4f*)(x + tid * 4);
    float ss = a[0]*a[0] + a[1]*a[1] + a[2]*a[2] + a[3]*a[3];
    float tot = block_sum(ss, sred);
    if (tid == 0) sInv = rsqrtf(tot * (1.f / DIM) + RMS_EPS);
    __syncthreads();
    float inv = sInv;
    float gp = 0.f;
    #pragma unroll
    for (int i = 0; i < 4; ++i) {
        float xi = a[i] * inv;
        Xn[(size_t)tok * DIM + tid * 4 + i] = f2bf(xi);
        gp += xi * Wgs[tid * 4 + i];
    }
    float g = block_sum(gp, sred);
    if (tid == 0) gate[tok] = 1.f / (1.f + expf(-g));
}

// ---------- generic bf16 WMMA GEMM: C = A(MxK) * B(NxK)^T ----------
// Register-blocked: each wave computes a 64x32 block of C (4x2 WMMA tiles),
// so each K-step loads 6 fragments and issues 8 WMMAs (~21 bf16-FLOP per byte
// of L2 traffic vs 8 for 1x1 tiling). OUT_BF16 selected at compile time so the
// epilogue is branch-free.
#define BM 4
#define BN 2
template <bool OUT_BF16>
__global__ void k_gemm_bf16(const unsigned short* __restrict__ A,
                            const unsigned short* __restrict__ B,
                            void* __restrict__ Cout,
                            int M, int N, int K) {
    const int lane = threadIdx.x & 31;
    const int wid  = threadIdx.x >> 5;
    int unit   = blockIdx.x * (blockDim.x >> 5) + wid;
    int unitsN = N / (16 * BN);
    int um = unit / unitsN, un = unit - um * unitsN;
    if (um >= M / (16 * BM)) return;
    int m0 = um * (16 * BM), n0 = un * (16 * BN);

    v8f acc[BM][BN];
    #pragma unroll
    for (int i = 0; i < BM; ++i)
        #pragma unroll
        for (int j = 0; j < BN; ++j) acc[i][j] = (v8f){};

    for (int k0 = 0; k0 < K; k0 += 32) {
        v16bf af[BM], bf[BN];
        #pragma unroll
        for (int i = 0; i < BM; ++i) af[i] = frag_a(A, K, m0 + 16 * i, k0, lane);
        #pragma unroll
        for (int j = 0; j < BN; ++j) bf[j] = frag_b(B, K, n0 + 16 * j, k0, lane);
        #pragma unroll
        for (int i = 0; i < BM; ++i)
            #pragma unroll
            for (int j = 0; j < BN; ++j)
                acc[i][j] = __builtin_amdgcn_wmma_f32_16x16x32_bf16(
                    false, af[i], false, bf[j], (short)0, acc[i][j], false, false);
    }
    int row0 = m0 + ((lane >> 4) << 3);
    int col0 = n0 + (lane & 15);
    #pragma unroll
    for (int i = 0; i < BM; ++i)
        #pragma unroll
        for (int j = 0; j < BN; ++j)
            #pragma unroll
            for (int r = 0; r < 8; ++r) {
                size_t off = (size_t)(row0 + 16 * i + r) * N + (col0 + 16 * j);
                if (OUT_BF16) ((unsigned short*)Cout)[off] = f2bf(acc[i][j][r]);
                else          ((float*)Cout)[off]          = acc[i][j][r];
            }
}

// ---------- kernel 5: per (token, half*head) q squared norm from bf16 q ----------
// one wave per (token, hh): 4 halves per lane via one 8-byte load + shuffle reduce
__global__ void k_q_post(const unsigned short* __restrict__ qbf, float* __restrict__ qs) {
    int lane = threadIdx.x & 31, wid = threadIdx.x >> 5;
    int id = blockIdx.x * 8 + wid;                    // token*8 + hh, 32768 waves
    const unsigned short* p = qbf + (size_t)(id >> 3) * DIM + (id & 7) * DQK + lane * 4;
    float s = 0.f;
    #pragma unroll
    for (int i = 0; i < 4; ++i) { float v = bf2f(p[i]); s += v * v; }
    s = wave_sum(s);
    if (lane == 0) qs[id] = s;
}

// ---------- kernel 6: fused q*k cross WMMA + top-8 per (token, half, head) ----------
__global__ void __launch_bounds__(1024)
k_dist_topk(const unsigned short* __restrict__ qbf, const unsigned short* __restrict__ kbf,
            const float* __restrict__ qs, const float* __restrict__ kss,
            float* __restrict__ sbuf, int* __restrict__ idxbuf) {
    __shared__ float sd[16][NUM_KEYS];                // 32 KB: 16 tokens x 512 keys
    const int lane = threadIdx.x & 31;
    const int w    = threadIdx.x >> 5;                // 32 waves
    const int hh   = blockIdx.y;                      // half*4 + head
    const int m0   = blockIdx.x << 4;                 // 16-token tile

    const unsigned short* A = qbf + hh * DQK;         // lda = 1024
    const unsigned short* B = kbf + (size_t)hh * NUM_KEYS * DQK; // ldb = 128
    int n0 = w << 4;                                  // each wave: 16 keys
    v8f acc = {};
    #pragma unroll
    for (int k0 = 0; k0 < DQK; k0 += 32) {
        v16bf af = frag_a(A, DIM, m0, k0, lane);
        v16bf bf = frag_b(B, DQK, n0, k0, lane);
        acc = __builtin_amdgcn_wmma_f32_16x16x32_bf16(false, af, false, bf,
                                                      (short)0, acc, false, false);
    }
    int tl0 = (lane >> 4) << 3;
    int key = n0 + (lane & 15);
    #pragma unroll
    for (int r = 0; r < 8; ++r) {
        int tl = tl0 + r;
        int tok = m0 + tl;
        sd[tl][key] = qs[tok * 8 + hh] + kss[hh * NUM_KEYS + key] - 2.f * acc[r] + IDW_EPS;
    }
    __syncthreads();

    if (w < 16) {                                     // wave w selects top-8 for token m0+w
        int tok = m0 + w;
        float loc[16];
        #pragma unroll
        for (int t = 0; t < 16; ++t) loc[t] = sd[w][t * 32 + lane];
        for (int r = 0; r < TOPK; ++r) {
            float bv = BIGF; int bk = 1 << 30;
            #pragma unroll
            for (int t = 0; t < 16; ++t)
                if (loc[t] < bv) { bv = loc[t]; bk = t * 32 + lane; }
            #pragma unroll
            for (int off = 16; off; off >>= 1) {
                float ov = __shfl_xor(bv, off, 32);
                int   ok = __shfl_xor(bk, off, 32);
                if (ov < bv || (ov == bv && ok < bk)) { bv = ov; bk = ok; }
            }
            if ((bk & 31) == lane) loc[bk >> 5] = BIGF;
            if (lane == 0) {
                size_t o = ((size_t)tok * 8 + hh) * TOPK + r;
                sbuf[o] = bv; idxbuf[o] = bk;
            }
        }
    }
}

// ---------- kernel 7: combine halves, top-8 of 64, softmax, gather, gate ----------
__global__ void k_combine_gather(const float* __restrict__ sbuf, const int* __restrict__ idxbuf,
                                 const float* __restrict__ memories, const float* __restrict__ tv,
                                 const float* __restrict__ gate, float* __restrict__ Ovec) {
    int lane = threadIdx.x & 31, wid = threadIdx.x >> 5;
    int unit = blockIdx.x * 8 + wid;                  // (token, head)
    int token = unit >> 2, head = unit & 3;

    float s1[TOPK], s2[TOPK]; int i1[TOPK], i2[TOPK];
    size_t b1 = ((size_t)token * 8 + head) * TOPK;
    size_t b2 = ((size_t)token * 8 + 4 + head) * TOPK;
    #pragma unroll
    for (int r = 0; r < TOPK; ++r) {
        s1[r] = sbuf[b1 + r]; i1[r] = idxbuf[b1 + r];
        s2[r] = sbuf[b2 + r]; i2[r] = idxbuf[b2 + r];
    }
    // two of the 64 candidate combos per lane: c = lane and lane+32 (c = i*8+j)
    float pr[2]; int ci[2] = { lane, lane + 32 };
    pr[0] = s1[lane >> 3]       * s2[lane & 7];
    pr[1] = s1[4 + (lane >> 3)] * s2[lane & 7];

    int selc[TOPK]; float selp[TOPK];
    for (int r = 0; r < TOPK; ++r) {
        float bv = BIGF; int bc = 1 << 30;
        #pragma unroll
        for (int u = 0; u < 2; ++u)
            if (pr[u] < bv || (pr[u] == bv && ci[u] < bc)) { bv = pr[u]; bc = ci[u]; }
        #pragma unroll
        for (int off = 16; off; off >>= 1) {
            float ov = __shfl_xor(bv, off, 32);
            int   oc = __shfl_xor(bc, off, 32);
            if (ov < bv || (ov == bv && oc < bc)) { bv = ov; bc = oc; }
        }
        selc[r] = bc; selp[r] = bv;
        #pragma unroll
        for (int u = 0; u < 2; ++u) if (ci[u] == bc) pr[u] = BIGF;
    }
    // softmax over scores = -log(prod)
    float m = -BIGF, wr[TOPK], wsum = 0.f;
    #pragma unroll
    for (int r = 0; r < TOPK; ++r) { float sc = -logf(selp[r]); wr[r] = sc; m = fmaxf(m, sc); }
    #pragma unroll
    for (int r = 0; r < TOPK; ++r) { wr[r] = expf(wr[r] - m); wsum += wr[r]; }
    float winv = 1.f / wsum;

    // gather memories rows and weighted-sum (4 floats per lane)
    v4f acc = {0.f, 0.f, 0.f, 0.f};
    #pragma unroll
    for (int r = 0; r < TOPK; ++r) {
        int c = selc[r];
        size_t mi = (size_t)i1[c >> 3] * NUM_KEYS + i2[c & 7];
        const v4f v = *(const v4f*)(memories + (mi * HEADS + head) * DV + lane * 4);
        float wk = wr[r] * winv;
        acc += wk * v;
    }
    // standardize tv head slice (128 elems across the wave)
    v4f t = *(const v4f*)(tv + (size_t)token * (HEADS * DV) + head * DV + lane * 4);
    float sm = wave_sum(t[0] + t[1] + t[2] + t[3]);
    float mean = sm * (1.f / 128.f);
    float sq = wave_sum(t[0]*t[0] + t[1]*t[1] + t[2]*t[2] + t[3]*t[3]);
    float var = (sq - 128.f * mean * mean) * (1.f / 127.f);
    float denom = fmaxf(sqrtf(fmaxf(var, 0.f)), 1e-10f);
    float g = gate[token];
    v4f out;
    #pragma unroll
    for (int i = 0; i < 4; ++i) {
        float ts = (t[i] - mean) / denom;
        out[i] = ts + g * (acc[i] - ts);
    }
    *(v4f*)(Ovec + (size_t)token * (HEADS * DV) + head * DV + lane * 4) = out;
}

// ---------- kernel 8: output rmsnorm -> bf16 ----------
__global__ void k_o_norm(const float* __restrict__ Ovec, unsigned short* __restrict__ On) {
    __shared__ float sred[4];
    __shared__ float sInv;
    int tok = blockIdx.x, tid = threadIdx.x;          // 128 threads x 4 elems
    v4f a = *(const v4f*)(Ovec + (size_t)tok * 512 + tid * 4);
    float ss = a[0]*a[0] + a[1]*a[1] + a[2]*a[2] + a[3]*a[3];
    float tot = block_sum(ss, sred);
    if (tid == 0) sInv = rsqrtf(tot * (1.f / 512.f) + RMS_EPS);
    __syncthreads();
    float inv = sInv;
    #pragma unroll
    for (int i = 0; i < 4; ++i)
        On[(size_t)tok * 512 + tid * 4 + i] = f2bf(a[i] * inv);
}

// ---------- host ----------
extern "C" void kernel_launch(void* const* d_in, const int* in_sizes, int n_in,
                              void* d_out, int out_size, void* d_ws, size_t ws_size,
                              hipStream_t stream) {
    (void)in_sizes; (void)n_in; (void)out_size; (void)ws_size;
    const float* tokens   = (const float*)d_in[0];
    const float* memories = (const float*)d_in[1];
    const float* keys     = (const float*)d_in[2];
    const float* q_norm_w = (const float*)d_in[3];
    const float* Wq       = (const float*)d_in[4];
    const float* g_norm_w = (const float*)d_in[5];
    const float* Wg       = (const float*)d_in[6];
    const float* v_norm_w = (const float*)d_in[7];
    const float* Wv       = (const float*)d_in[8];
    const float* o_norm_w = (const float*)d_in[9];
    const float* Wo       = (const float*)d_in[10];
    float* out = (float*)d_out;

    char* p = (char*)d_ws;
    auto alloc = [&](size_t bytes) -> void* {
        void* r = (void*)p;
        p += (bytes + 255) & ~(size_t)255;
        return r;
    };
    unsigned short* Xn   = (unsigned short*)alloc((size_t)NTOK * DIM * 2);   // 8 MB
    unsigned short* qbf  = (unsigned short*)alloc((size_t)NTOK * DIM * 2);   // 8 MB
    float*          tv   = (float*)alloc((size_t)NTOK * 512 * 4);            // 8 MB
    float*          Ovec = (float*)alloc((size_t)NTOK * 512 * 4);            // 8 MB
    unsigned short* On   = (unsigned short*)alloc((size_t)NTOK * 512 * 2);   // 4 MB
    unsigned short* Wqb  = (unsigned short*)alloc((size_t)1024 * 1024 * 2);  // 2 MB
    unsigned short* Wvb  = (unsigned short*)alloc((size_t)512 * 1024 * 2);   // 1 MB
    unsigned short* Wob  = (unsigned short*)alloc((size_t)1024 * 512 * 2);   // 1 MB
    unsigned short* Kb   = (unsigned short*)alloc((size_t)8 * 512 * 128 * 2);// 1 MB
    float*          Wgs  = (float*)alloc(1024 * 4);
    float*          kssb = (float*)alloc((size_t)8 * 512 * 4);
    float*          gateb= (float*)alloc((size_t)NTOK * 4);
    float*          qsb  = (float*)alloc((size_t)NTOK * 8 * 4);
    float*          sb   = (float*)alloc((size_t)NTOK * 8 * TOPK * 4);       // 1 MB
    int*            ib   = (int*)alloc((size_t)NTOK * 8 * TOPK * 4);         // 1 MB

    // 1) fold norm weights, cast to bf16
    {
        int total = 1024 * 1024 + 512 * 1024 + 1024 * 512 + 8 * 512 * 128 + 1024;
        k_prep_weights<<<(total + 255) / 256, 256, 0, stream>>>(
            Wq, q_norm_w, Wv, v_norm_w, Wo, o_norm_w, keys, Wg, g_norm_w,
            Wqb, Wvb, Wob, Kb, Wgs);
    }
    // 2) key squared norms
    k_kss<<<16, 256, 0, stream>>>(keys, kssb);
    // 3) token rmsnorm (shared across q/g/v) + gate
    k_norm_tokens<<<NTOK, 256, 0, stream>>>(tokens, Wgs, Xn, gateb);
    // 4) q = Xn @ Wq'^T  (bf16 out): (4096/64)*(1024/32)=2048 wave units /8
    k_gemm_bf16<true><<<256, 256, 0, stream>>>(Xn, Wqb, qbf, NTOK, 1024, 1024);
    // 5) q squared norms (one wave per (token,hh))
    k_q_post<<<(NTOK * 8) / 8, 256, 0, stream>>>(qbf, qsb);
    // 6) tv = Xn @ Wv'^T  (f32 out): (4096/64)*(512/32)=1024 units /8
    k_gemm_bf16<false><<<128, 256, 0, stream>>>(Xn, Wvb, tv, NTOK, 512, 1024);
    // 7) fused distances + per-(half,head) top-8
    k_dist_topk<<<dim3(NTOK / 16, 8), 1024, 0, stream>>>(qbf, Kb, qsb, kssb, sb, ib);
    // 8) combine halves, select 8 of 64, softmax, gather memories, gate with tv
    k_combine_gather<<<(NTOK * HEADS) / 8, 256, 0, stream>>>(
        sb, ib, memories, tv, gateb, Ovec);
    // 9) output rmsnorm -> bf16
    k_o_norm<<<NTOK, 128, 0, stream>>>(Ovec, On);
    // 10) out = On @ Wo'^T -> d_out (f32): 2048 units /8
    k_gemm_bf16<false><<<256, 256, 0, stream>>>(On, Wob, out, NTOK, 1024, 512);
}
